// FSEncoderSized_34119220199466
// MI455X (gfx1250) — compile-verified, matrix operand order checked by reading, not run on hardware
//
#include <hip/hip_runtime.h>

// ---- vector types (plain ext_vector so unions stay POD) ----
typedef __attribute__((ext_vector_type(16))) _Float16 v16h;
typedef __attribute__((ext_vector_type(8)))  float    v8f;
typedef __attribute__((ext_vector_type(4)))  unsigned int v4u;
typedef __attribute__((ext_vector_type(4)))  float    v4f;

union F16x16 { v16h v; v4u q[2]; _Float16 h[16]; };
union F16x8  { v4u  q; _Float16 h[8]; };

#define DIM    256
#define COUT   128
#define NTOT   512
#define CHUNK  128
#define SPAD   264   // padded f16 stride for activation buffers [n][k]
#define HSTR   136   // padded f16 stride for final h map [n][c]

// dynamic LDS layout (bytes)
#define OFF_BUFA 0
#define SZ_BUF   (CHUNK * SPAD * 2)            // 67584
#define OFF_BUFB (OFF_BUFA + SZ_BUF)           // 67584
#define OFF_HFIN (OFF_BUFB + SZ_BUF)           // 135168
#define SZ_HFIN  (NTOT * HSTR * 2)             // 139264
#define OFF_X    (OFF_HFIN + SZ_HFIN)          // 274432
#define SZ_X     (5 * CHUNK * 4)               // 2560
#define OFF_MASK (OFF_X + SZ_X)                // 276992
#define SMEM_BYTES (OFF_MASK + NTOT * 4)       // 279040  (< 320KB/WGP)

// ---- prep kernels ----
__global__ void cvt_f32_f16(const float* __restrict__ s, _Float16* __restrict__ d, int n) {
    int i = blockIdx.x * 256 + threadIdx.x;
    if (i < n) d[i] = (_Float16)s[i];
}

// wtab[c][n] = (1-frac)*pool_w[c,idx] + frac*pool_w[c,min(idx+1,20)], pos = 20*n/511
__global__ void build_wtab(const float* __restrict__ pw, float* __restrict__ wtab) {
    int i = blockIdx.x * 256 + threadIdx.x;   // 128*512 = 65536 total
    int c = i >> 9, n = i & 511;
    float pos  = 20.0f * ((float)n / 511.0f);
    int   idx  = (int)pos;                    // 0..20
    float frac = pos - (float)idx;
    int   idr  = (idx + 1 > 20) ? 20 : idx + 1;
    wtab[c * 512 + n] = (1.0f - frac) * pw[c * 21 + idx] + frac * pw[c * 21 + idr];
}

// ---- one wave computes one 16-row output tile band (all 8 n-tiles) of a dim=256 layer ----
__device__ __forceinline__ void gemm_tile_row(
    const _Float16* __restrict__ wgt,   // f16 weights, row-major [M][256]
    int mt,                             // M-tile index (M = mt*16 .. +15)
    const _Float16* inb,                // LDS activations [n][k], stride SPAD
    const float* __restrict__ bias,
    _Float16* outb,                     // LDS output [n][k], stride SPAD
    int lane)
{
    F16x16 a[8];
    const int mrow = mt * 16 + (lane & 15);
    const int koff = (lane >> 4) * 8;                 // hi half-wave: K+8
    const v4u* wp = (const v4u*)(wgt + mrow * DIM + koff);
    #pragma unroll
    for (int kt = 0; kt < 8; ++kt) {                  // A frag: K = kt*32 + koff + {0..7, 16..23}
        a[kt].q[0] = wp[kt * 4];
        a[kt].q[1] = wp[kt * 4 + 2];
    }
    const int m0 = mt * 16 + koff;
    v4f blo = *(const v4f*)(bias + m0);
    v4f bhi = *(const v4f*)(bias + m0 + 4);
    float bb[8] = {blo.x, blo.y, blo.z, blo.w, bhi.x, bhi.y, bhi.z, bhi.w};

    #pragma unroll 1
    for (int nt = 0; nt < 8; ++nt) {
        v8f acc = {0.f, 0.f, 0.f, 0.f, 0.f, 0.f, 0.f, 0.f};
        const _Float16* bp = inb + (nt * 16 + (lane & 15)) * SPAD + koff;
        #pragma unroll
        for (int kt = 0; kt < 8; ++kt) {
            F16x16 bf;
            bf.q[0] = *(const v4u*)(bp + kt * 32);
            bf.q[1] = *(const v4u*)(bp + kt * 32 + 16);
            acc = __builtin_amdgcn_wmma_f32_16x16x32_f16(
                false, a[kt].v, false, bf.v, (short)0, acc, false, false);
        }
        F16x8 st;   // D lane layout: element r -> (M = mt*16 + koff + r, N = nt*16 + lane&15)
        #pragma unroll
        for (int r = 0; r < 8; ++r)
            st.h[r] = (_Float16)fmaxf(acc[r] + bb[r], 0.0f);
        *(v4u*)(outb + (nt * 16 + (lane & 15)) * SPAD + mt * 16 + koff) = st.q;
    }
}

// ---- fused encoder: one workgroup (8 wave32) per batch element ----
__global__ void __launch_bounds__(256)
fsenc_main(const float* __restrict__ x, const float* __restrict__ mask,
           const float* __restrict__ w1, const float* __restrict__ b1,
           const float* __restrict__ b2, const float* __restrict__ b3,
           const float* __restrict__ b4,
           const _Float16* __restrict__ w2h, const _Float16* __restrict__ w3h,
           const _Float16* __restrict__ w4h, const float* __restrict__ wtab,
           float* __restrict__ out)
{
    extern __shared__ char smem[];
    _Float16* bufA    = (_Float16*)(smem + OFF_BUFA);
    _Float16* bufB    = (_Float16*)(smem + OFF_BUFB);
    _Float16* hfin    = (_Float16*)(smem + OFF_HFIN);   // [n][c] f16, stride HSTR
    float*    x_lds   = (float*)(smem + OFF_X);
    float*    mask_lds= (float*)(smem + OFF_MASK);
    float*    rowbuf  = (float*)(smem);                 // aliases bufA after conv chain

    const int b    = blockIdx.x;
    const int tid  = threadIdx.x;
    const int lane = tid & 31;
    const int wave = tid >> 5;

    for (int i = tid; i < NTOT; i += 256) mask_lds[i] = mask[b * NTOT + i];

    for (int ch = 0; ch < 4; ++ch) {
        const int nbase = ch * CHUNK;
        for (int i = tid; i < 5 * CHUNK; i += 256)
            x_lds[i] = x[b * 5 * NTOT + (i >> 7) * NTOT + nbase + (i & 127)];
        __syncthreads();

        // ---- layer 1 (K=5, scalar VALU): h1 = relu(w1 x + b1), write [n][k] f16 ----
        {
            const int n = tid & 127, half = tid >> 7;
            float xv[5];
            #pragma unroll
            for (int c = 0; c < 5; ++c) xv[c] = x_lds[c * CHUNK + n];
            for (int g = 0; g < 16; ++g) {
                int d0 = (g * 2 + half) * 8;
                F16x8 st;
                #pragma unroll
                for (int r = 0; r < 8; ++r) {
                    int d = d0 + r;
                    float acc = b1[d];
                    #pragma unroll
                    for (int c = 0; c < 5; ++c) acc += w1[d * 5 + c] * xv[c];
                    st.h[r] = (_Float16)fmaxf(acc, 0.0f);
                }
                *(v4u*)(bufA + n * SPAD + d0) = st.q;
            }
        }
        __syncthreads();

        // ---- layers 2,3: 256x256 GEMMs via v_wmma_f32_16x16x32_f16 ----
        gemm_tile_row(w2h, wave * 2 + 0, bufA, b2, bufB, lane);
        gemm_tile_row(w2h, wave * 2 + 1, bufA, b2, bufB, lane);
        __syncthreads();
        gemm_tile_row(w3h, wave * 2 + 0, bufB, b3, bufA, lane);
        gemm_tile_row(w3h, wave * 2 + 1, bufB, b3, bufA, lane);
        __syncthreads();

        // ---- layer 4: 128x256, epilogue = (+b4)/512 * mask, store into hfin [n][c] ----
        {
            const int mt = wave;
            F16x16 a[8];
            const int mrow = mt * 16 + (lane & 15);
            const int koff = (lane >> 4) * 8;
            const v4u* wp = (const v4u*)(w4h + mrow * DIM + koff);
            #pragma unroll
            for (int kt = 0; kt < 8; ++kt) { a[kt].q[0] = wp[kt * 4]; a[kt].q[1] = wp[kt * 4 + 2]; }
            const int m0 = mt * 16 + koff;
            v4f blo = *(const v4f*)(b4 + m0);
            v4f bhi = *(const v4f*)(b4 + m0 + 4);
            float bb[8] = {blo.x, blo.y, blo.z, blo.w, bhi.x, bhi.y, bhi.z, bhi.w};
            #pragma unroll 1
            for (int nt = 0; nt < 8; ++nt) {
                v8f acc = {0.f, 0.f, 0.f, 0.f, 0.f, 0.f, 0.f, 0.f};
                const _Float16* bp = bufA + (nt * 16 + (lane & 15)) * SPAD + koff;
                #pragma unroll
                for (int kt = 0; kt < 8; ++kt) {
                    F16x16 bf;
                    bf.q[0] = *(const v4u*)(bp + kt * 32);
                    bf.q[1] = *(const v4u*)(bp + kt * 32 + 16);
                    acc = __builtin_amdgcn_wmma_f32_16x16x32_f16(
                        false, a[kt].v, false, bf.v, (short)0, acc, false, false);
                }
                const int ncol = nbase + nt * 16 + (lane & 15);
                const float mval = mask_lds[ncol] * (1.0f / 512.0f);
                F16x8 st;
                #pragma unroll
                for (int r = 0; r < 8; ++r)
                    st.h[r] = (_Float16)((acc[r] + bb[r]) * mval);
                *(v4u*)(hfin + ncol * HSTR + mt * 16 + koff) = st.q;
            }
        }
        __syncthreads();
    }

    // ---- FSPool: each wave sorts one channel row (descending) and dots with wtab ----
    float* myrow = rowbuf + wave * 512;
    for (int t = 0; t < 16; ++t) {
        const int c = t * 8 + wave;
        for (int i = lane; i < 512; i += 32)
            myrow[i] = (float)hfin[i * HSTR + c];
        __syncthreads();
        for (int k = 2; k <= 512; k <<= 1) {
            for (int j = k >> 1; j > 0; j >>= 1) {
                for (int i = lane; i < 512; i += 32) {
                    int ixj = i ^ j;
                    if (ixj > i) {
                        float av = myrow[i], bv = myrow[ixj];
                        bool desc = ((i & k) == 0);
                        if ((av < bv) == desc) { myrow[i] = bv; myrow[ixj] = av; }
                    }
                }
                __syncthreads();
            }
        }
        float s = 0.0f;
        for (int i = lane; i < 512; i += 32)
            s += myrow[i] * wtab[c * 512 + i];
        #pragma unroll
        for (int off = 16; off > 0; off >>= 1) s += __shfl_xor(s, off, 32);
        if (lane == 0) out[b * 129 + c] = s;
        __syncthreads();
    }

    // ---- size feature: mean(mask)*4 ----
    if (wave == 0) {
        float s = 0.0f;
        for (int i = lane; i < 512; i += 32) s += mask_lds[i];
        #pragma unroll
        for (int off = 16; off > 0; off >>= 1) s += __shfl_xor(s, off, 32);
        if (lane == 0) out[b * 129 + 128] = s * (4.0f / 512.0f);
    }
}

extern "C" void kernel_launch(void* const* d_in, const int* in_sizes, int n_in,
                              void* d_out, int out_size, void* d_ws, size_t ws_size,
                              hipStream_t stream) {
    const float* x    = (const float*)d_in[0];
    const float* mask = (const float*)d_in[1];
    const float* w1   = (const float*)d_in[2];
    const float* b1   = (const float*)d_in[3];
    const float* w2   = (const float*)d_in[4];
    const float* b2   = (const float*)d_in[5];
    const float* w3   = (const float*)d_in[6];
    const float* b3   = (const float*)d_in[7];
    const float* w4   = (const float*)d_in[8];
    const float* b4   = (const float*)d_in[9];
    const float* pw   = (const float*)d_in[10];
    float* out = (float*)d_out;

    char* ws = (char*)d_ws;
    _Float16* w2h  = (_Float16*)(ws + 0);        // 256*256 f16 = 128KB
    _Float16* w3h  = (_Float16*)(ws + 131072);   // 128KB
    _Float16* w4h  = (_Float16*)(ws + 262144);   // 128*256 f16 = 64KB
    float*    wtab = (float*)(ws + 327680);      // 128*512 f32 = 256KB

    cvt_f32_f16<<<256, 256, 0, stream>>>(w2, w2h, 65536);
    cvt_f32_f16<<<256, 256, 0, stream>>>(w3, w3h, 65536);
    cvt_f32_f16<<<128, 256, 0, stream>>>(w4, w4h, 32768);
    build_wtab<<<256, 256, 0, stream>>>(pw, wtab);

    fsenc_main<<<1024, 256, SMEM_BYTES, stream>>>(
        x, mask, w1, b1, b2, b3, b4, w2h, w3h, w4h, wtab, out);
}
